// scatterMoE_49486613184898
// MI455X (gfx1250) — compile-verified, matrix operand order checked by reading, not run on hardware
//
#include <hip/hip_runtime.h>
#include <hip/hip_bf16.h>
#include <math.h>

// ---------------------------------------------------------------------------
// Problem constants (B,S,D,E,K = 4,2048,1024,8,2)
// ---------------------------------------------------------------------------
#define T_TOK 8192      // B*S tokens
#define D_DIM 1024      // hidden dim
#define H_DIM 4096      // FFN dim (4*D)
#define E_NUM 8         // experts
#define TOPK  2

// LDS row strides padded so row-to-row bank delta is 4 dwords (mod 64):
// lanes of a half-wave fan out over all 64 banks instead of hitting one.
#define SX_STRIDE   1032    // bf16 elems (2064 B, 16B-aligned rows)
#define SACT_STRIDE 264     // bf16 elems ( 528 B, 16B-aligned rows)

typedef __bf16 bf16_t;
typedef bf16_t v16bf __attribute__((ext_vector_type(16)));
typedef bf16_t v8bf  __attribute__((ext_vector_type(8)));
typedef bf16_t v2bf  __attribute__((ext_vector_type(2)));
typedef float  v8f   __attribute__((ext_vector_type(8)));

static __device__ __forceinline__ v16bf cat8(v8bf a, v8bf b) {
    return __builtin_shufflevector(a, b, 0,1,2,3,4,5,6,7,8,9,10,11,12,13,14,15);
}
static __device__ __forceinline__ float gelu_exact(float v) {
    return 0.5f * v * (1.0f + erff(v * 0.70710678118654752f));
}

// ===========================================================================
// Kernel 1: router. One wave32 per token.
// ===========================================================================
__global__ __launch_bounds__(256) void router_kernel(
    const float* __restrict__ x, const float* __restrict__ gate_w,
    float* __restrict__ logits_out, int* __restrict__ sel,
    float* __restrict__ wts)
{
    const int t = blockIdx.x * (blockDim.x >> 5) + (threadIdx.x >> 5);
    const int lane = threadIdx.x & 31;
    if (t >= T_TOK) return;

    const float* xr = x + (size_t)t * D_DIM;
    float acc[E_NUM];
#pragma unroll
    for (int e = 0; e < E_NUM; ++e) acc[e] = 0.0f;
    for (int i = lane; i < D_DIM; i += 32) {
        const float xv = xr[i];
        const float* g = gate_w + (size_t)i * E_NUM;
#pragma unroll
        for (int e = 0; e < E_NUM; ++e) acc[e] += xv * g[e];
    }
#pragma unroll
    for (int e = 0; e < E_NUM; ++e)
        for (int off = 16; off >= 1; off >>= 1)
            acc[e] += __shfl_xor(acc[e], off, 32);

    if (lane == 0) {
        float m = acc[0];
#pragma unroll
        for (int e = 1; e < E_NUM; ++e) m = fmaxf(m, acc[e]);
        float p[E_NUM], s = 0.0f;
#pragma unroll
        for (int e = 0; e < E_NUM; ++e) { p[e] = __expf(acc[e] - m); s += p[e]; }
        const float inv = 1.0f / s;
#pragma unroll
        for (int e = 0; e < E_NUM; ++e) {
            p[e] *= inv;
            logits_out[(size_t)t * E_NUM + e] = acc[e];
        }
        int i0 = 0;
#pragma unroll
        for (int e = 1; e < E_NUM; ++e) if (p[e] > p[i0]) i0 = e;
        int i1 = (i0 == 0) ? 1 : 0;
#pragma unroll
        for (int e = 0; e < E_NUM; ++e) if (e != i0 && p[e] > p[i1]) i1 = e;
        const float sw = 1.0f / (p[i0] + p[i1]);
        sel[2 * t + 0] = i0;  wts[2 * t + 0] = p[i0] * sw;
        sel[2 * t + 1] = i1;  wts[2 * t + 1] = p[i1] * sw;
    }
}

// ===========================================================================
// Kernel 2: deterministic per-expert gather lists (ballot/popc compaction,
// order-stable). Pads to a multiple of 32 rows with weight-0 dummies.
// ===========================================================================
__global__ __launch_bounds__(256) void build_lists_kernel(
    const int* __restrict__ sel, const float* __restrict__ wts,
    int* __restrict__ listTok, float* __restrict__ listWt,
    int* __restrict__ ntile)
{
    const int w = threadIdx.x >> 5;       // expert id
    const int lane = threadIdx.x & 31;
    int cnt = 0;
    for (int j = lane; j < T_TOK * TOPK; j += 32) {
        const int s = sel[j];
        const bool m = (s == w);
        const unsigned int bal = (unsigned int)__ballot(m);
        const int rank = __popc(bal & ((1u << lane) - 1u));
        if (m) {
            listTok[w * T_TOK + cnt + rank] = j >> 1;
            listWt [w * T_TOK + cnt + rank] = wts[j];
        }
        cnt += __popc(bal);
    }
    const int padded = (cnt + 31) & ~31;
    for (int p = cnt + lane; p < padded; p += 32) {
        listTok[w * T_TOK + p] = 0;
        listWt [w * T_TOK + p] = 0.0f;
    }
    if (lane == 0) { ntile[w] = padded >> 4; ntile[8 + w] = padded >> 5; }
}

// ===========================================================================
// Kernel 3: zero the MoE output region.
// ===========================================================================
__global__ __launch_bounds__(256) void zero_kernel(float4* __restrict__ p, int n4)
{
    const int i = blockIdx.x * blockDim.x + threadIdx.x;
    if (i < n4) p[i] = make_float4(0.f, 0.f, 0.f, 0.f);
}

// ===========================================================================
// Weight swizzle kernels: fp32 [k][n] -> bf16 tiles in WMMA B-operand order.
// Tile = 32(k) x 16(n); per tile: [lane][16 elems] contiguous (1 KB), so a
// lane's whole operand is 32 contiguous bytes (2x global_load_b128).
// Lane L: n = (L&15), g = L>>4;  elem j<8 -> k = 8g+j ; j>=8 -> k = 16+8g+(j-8).
//   w1s tile order: ((e*256 + n0i)*32 + kbi)        (kb fastest = stream order)
//   w2s tile order: ((e*64  + ndi)*128 + hti)       (h  fastest = stream order)
// ===========================================================================
__global__ __launch_bounds__(256) void swz_w1_kernel(
    const float* __restrict__ w1, bf16_t* __restrict__ w1s)
{
    const int wid  = blockIdx.x * 8 + (threadIdx.x >> 5); // tile id
    const int lane = threadIdx.x & 31;
    const int lo = lane & 15, g = lane >> 4;
    const int kbi = wid & 31, n0i = (wid >> 5) & 255, e = wid >> 13;
    const float* src = w1 + ((size_t)e * D_DIM) * H_DIM + (size_t)(n0i * 16 + lo);
    bf16_t* dst = w1s + (size_t)wid * 512 + lane * 16;
#pragma unroll
    for (int j = 0; j < 8; ++j) {
        dst[j]     = (bf16_t)src[(size_t)(kbi * 32 + 8 * g + j) * H_DIM];
        dst[j + 8] = (bf16_t)src[(size_t)(kbi * 32 + 16 + 8 * g + j) * H_DIM];
    }
}

__global__ __launch_bounds__(256) void swz_w2_kernel(
    const float* __restrict__ w2, bf16_t* __restrict__ w2s)
{
    const int wid  = blockIdx.x * 8 + (threadIdx.x >> 5);
    const int lane = threadIdx.x & 31;
    const int lo = lane & 15, g = lane >> 4;
    const int hti = wid & 127, ndi = (wid >> 7) & 63, e = wid >> 13;
    const float* src = w2 + ((size_t)e * H_DIM) * D_DIM + (size_t)(ndi * 16 + lo);
    bf16_t* dst = w2s + (size_t)wid * 512 + lane * 16;
#pragma unroll
    for (int j = 0; j < 8; ++j) {
        dst[j]     = (bf16_t)src[(size_t)(hti * 32 + 8 * g + j) * D_DIM];
        dst[j + 8] = (bf16_t)src[(size_t)(hti * 32 + 16 + 8 * g + j) * D_DIM];
    }
}

// ===========================================================================
// Kernel 4 (fast path): fused expert FFN, 32-token tiles, 16 waves.
//   LDS: X tile (padded) 32xSX_STRIDE bf16 + act chunk 32xSACT_STRIDE bf16.
//   for hb in 0..4096 step 256:
//     stage1: wave w -> act[0:32, hb+16w..+16] = gelu(X @ W1)   K=1024
//     stage2: wave w -> out[0:32, 64w..+64]   += act @ W2       C in VGPRs
//   B tiles batch-loaded (4 tiles / 8 b128 loads in flight) so the compiler
//   drains them with progressive s_wait_loadcnt instead of wait-0 per tile.
// ===========================================================================
__global__ __launch_bounds__(512) void moe_ffn32_kernel(
    const float*  __restrict__ x,   const bf16_t* __restrict__ w1s,
    const bf16_t* __restrict__ w2s, const int*    __restrict__ listTok,
    const float*  __restrict__ listWt, const int* __restrict__ ntile,
    float* __restrict__ out)
{
    __shared__ bf16_t sX[32 * SX_STRIDE];      // ~64.5 KB
    __shared__ bf16_t sAct[32 * SACT_STRIDE];  // ~16.5 KB
    __shared__ int    sTok[32];
    __shared__ float  sWt[32];

    const int e = blockIdx.y;
    const int tile = blockIdx.x;
    if (tile >= ntile[8 + e]) return;

    const int tid  = threadIdx.x;
    const int wv   = tid >> 5;           // wave 0..15
    const int lane = tid & 31;
    const int lo   = lane & 15;
    const int g    = lane >> 4;

    if (tid < 32) {
        sTok[tid] = listTok[e * T_TOK + tile * 32 + tid];
        sWt [tid] = listWt [e * T_TOK + tile * 32 + tid];
    }
    __syncthreads();
    // Gather 32 token rows -> LDS bf16; float2 reads, packed b32 stores.
    for (int idx = tid; idx < 32 * 512; idx += 512) {
        const int r = idx >> 9, cp = idx & 511;
        const float2 xv = *(const float2*)(x + (size_t)sTok[r] * D_DIM + 2 * cp);
        *(v2bf*)&sX[r * SX_STRIDE + 2 * cp] = (v2bf){(bf16_t)xv.x, (bf16_t)xv.y};
    }
    __syncthreads();

    const bf16_t* W1 = w1s + (size_t)e * D_DIM * H_DIM;   // e*256*32*512
    const bf16_t* W2 = w2s + (size_t)e * H_DIM * D_DIM;   // e*64*128*512

    v8f o[2][4];
#pragma unroll
    for (int mb = 0; mb < 2; ++mb)
#pragma unroll
        for (int nt = 0; nt < 4; ++nt) o[mb][nt] = (v8f){0,0,0,0,0,0,0,0};

    const int c0 = wv * 16;    // stage1 act columns of this wave

    for (int hb = 0; hb < H_DIM; hb += 256) {
        // ---- stage 1: act[0:32, hb+c0 .. +16] = X @ W1, K = 1024
        v8f a[2] = { (v8f){0,0,0,0,0,0,0,0}, (v8f){0,0,0,0,0,0,0,0} };
        {
            const int n0i = (hb + c0) >> 4;
            const bf16_t* bb = W1 + ((size_t)n0i * 32) * 512 + lane * 16;
            for (int kq = 0; kq < 8; ++kq) {          // quads of k-tiles
                v8bf q0[4], q1[4];
#pragma unroll
                for (int u = 0; u < 4; ++u) {         // 8 loads in flight
                    q0[u] = *(const v8bf*)(bb + u * 512);
                    q1[u] = *(const v8bf*)(bb + u * 512 + 8);
                }
#pragma unroll
                for (int u = 0; u < 4; ++u) {
                    const int kbi = kq * 4 + u;
                    const v16bf B = cat8(q0[u], q1[u]);
#pragma unroll
                    for (int mb = 0; mb < 2; ++mb) {
                        const int m = lo + 16 * mb;
                        const v16bf A = cat8(
                            *(const v8bf*)&sX[m * SX_STRIDE + kbi * 32 + 8 * g],
                            *(const v8bf*)&sX[m * SX_STRIDE + kbi * 32 + 16 + 8 * g]);
                        a[mb] = __builtin_amdgcn_wmma_f32_16x16x32_bf16(
                            false, A, false, B, (short)0, a[mb], false, false);
                    }
                }
                bb += 4 * 512;
            }
        }
        __syncthreads();   // prior stage2 reads of sAct done
#pragma unroll
        for (int mb = 0; mb < 2; ++mb) {
#pragma unroll
            for (int v = 0; v < 8; ++v) {
                const int m = 16 * mb + v + 8 * g;
                sAct[m * SACT_STRIDE + c0 + lo] = (bf16_t)gelu_exact(a[mb][v]);
            }
        }
        __syncthreads();
        // ---- stage 2: out[0:32, 64w..+64] += act @ W2 over this 256 chunk
        for (int k2i = 0; k2i < 8; ++k2i) {
            const int hti = (hb >> 5) + k2i;
            // nt tiles are 128*512 = 65536 elems apart in w2s
            const bf16_t* base =
                W2 + ((size_t)(wv * 4) * 128 + hti) * 512 + lane * 16;
            v8bf q0[4], q1[4];
#pragma unroll
            for (int nt = 0; nt < 4; ++nt) {          // 8 loads in flight
                q0[nt] = *(const v8bf*)(base + (size_t)nt * 65536);
                q1[nt] = *(const v8bf*)(base + (size_t)nt * 65536 + 8);
            }
            v16bf A[2];
#pragma unroll
            for (int mb = 0; mb < 2; ++mb) {
                const int m = lo + 16 * mb;
                A[mb] = cat8(
                    *(const v8bf*)&sAct[m * SACT_STRIDE + k2i * 32 + 8 * g],
                    *(const v8bf*)&sAct[m * SACT_STRIDE + k2i * 32 + 16 + 8 * g]);
            }
#pragma unroll
            for (int nt = 0; nt < 4; ++nt) {
                const v16bf B = cat8(q0[nt], q1[nt]);
                o[0][nt] = __builtin_amdgcn_wmma_f32_16x16x32_bf16(
                    false, A[0], false, B, (short)0, o[0][nt], false, false);
                o[1][nt] = __builtin_amdgcn_wmma_f32_16x16x32_bf16(
                    false, A[1], false, B, (short)0, o[1][nt], false, false);
            }
        }
    }

    // Scatter-add: row m = 16*mb + v + 8g, col = 64*wv + 16*nt + lo.
#pragma unroll
    for (int mb = 0; mb < 2; ++mb) {
        int tk[8]; float tw[8];
#pragma unroll
        for (int v = 0; v < 8; ++v) {
            tk[v] = sTok[16 * mb + v + 8 * g];
            tw[v] = sWt [16 * mb + v + 8 * g];
        }
#pragma unroll
        for (int nt = 0; nt < 4; ++nt)
#pragma unroll
            for (int v = 0; v < 8; ++v)
                atomicAdd(out + (size_t)tk[v] * D_DIM + wv * 64 + nt * 16 + lo,
                          o[mb][nt][v] * tw[v]);
    }
}

// ===========================================================================
// Fallback FFN: 16-token tiles, direct fp32 weight loads (small-ws case).
// ===========================================================================
__global__ __launch_bounds__(256) void moe_ffn_kernel(
    const float* __restrict__ x,  const float* __restrict__ w1,
    const float* __restrict__ w2, const int*   __restrict__ listTok,
    const float* __restrict__ listWt, const int* __restrict__ ntile,
    float* __restrict__ out)
{
    __shared__ bf16_t sX[16 * SX_STRIDE];
    __shared__ bf16_t sAct[16 * SACT_STRIDE];
    __shared__ int    sTok[16];
    __shared__ float  sWt[16];

    const int e = blockIdx.y;
    const int tile = blockIdx.x;
    if (tile >= ntile[e]) return;

    const int tid  = threadIdx.x;
    const int wv   = tid >> 5;
    const int lane = tid & 31;
    const int lo   = lane & 15;
    const int g    = lane >> 4;

    if (tid < 16) {
        sTok[tid] = listTok[e * T_TOK + tile * 16 + tid];
        sWt [tid] = listWt [e * T_TOK + tile * 16 + tid];
    }
    __syncthreads();
    for (int idx = tid; idx < 16 * 512; idx += 256) {
        const int r = idx >> 9, cp = idx & 511;
        const float2 xv = *(const float2*)(x + (size_t)sTok[r] * D_DIM + 2 * cp);
        *(v2bf*)&sX[r * SX_STRIDE + 2 * cp] = (v2bf){(bf16_t)xv.x, (bf16_t)xv.y};
    }
    __syncthreads();

    const float* W1 = w1 + (size_t)e * D_DIM * H_DIM;
    const float* W2 = w2 + (size_t)e * H_DIM * D_DIM;

    v8f o[8];
#pragma unroll
    for (int nt = 0; nt < 8; ++nt) o[nt] = (v8f){0,0,0,0,0,0,0,0};

    const int c0  = wv * 32;
    const int oc0 = wv * 128;

    for (int hb = 0; hb < H_DIM; hb += 256) {
        v8f a0 = (v8f){0,0,0,0,0,0,0,0};
        v8f a1 = (v8f){0,0,0,0,0,0,0,0};
        for (int kb = 0; kb < D_DIM; kb += 32) {
            const v16bf A = cat8(*(const v8bf*)&sX[lo * SX_STRIDE + kb + 8 * g],
                                 *(const v8bf*)&sX[lo * SX_STRIDE + kb + 16 + 8 * g]);
            const float* bp = W1 + (size_t)kb * H_DIM + (hb + c0) + lo;
            v16bf B0, B1;
#pragma unroll
            for (int j = 0; j < 8; ++j) {
                B0[j]     = (bf16_t)bp[(size_t)(8 * g + j) * H_DIM];
                B0[j + 8] = (bf16_t)bp[(size_t)(16 + 8 * g + j) * H_DIM];
                B1[j]     = (bf16_t)bp[(size_t)(8 * g + j) * H_DIM + 16];
                B1[j + 8] = (bf16_t)bp[(size_t)(16 + 8 * g + j) * H_DIM + 16];
            }
            a0 = __builtin_amdgcn_wmma_f32_16x16x32_bf16(false, A, false, B0,
                                                         (short)0, a0, false, false);
            a1 = __builtin_amdgcn_wmma_f32_16x16x32_bf16(false, A, false, B1,
                                                         (short)0, a1, false, false);
        }
        __syncthreads();
#pragma unroll
        for (int v = 0; v < 8; ++v) {
            const int m = v + 8 * g;
            sAct[m * SACT_STRIDE + c0 + lo]      = (bf16_t)gelu_exact(a0[v]);
            sAct[m * SACT_STRIDE + c0 + 16 + lo] = (bf16_t)gelu_exact(a1[v]);
        }
        __syncthreads();
        for (int k2 = 0; k2 < 256; k2 += 32) {
            const v16bf A = cat8(*(const v8bf*)&sAct[lo * SACT_STRIDE + k2 + 8 * g],
                                 *(const v8bf*)&sAct[lo * SACT_STRIDE + k2 + 16 + 8 * g]);
            const float* b2 = W2 + (size_t)(hb + k2) * D_DIM + oc0 + lo;
#pragma unroll
            for (int nt = 0; nt < 8; ++nt) {
                const float* bp = b2 + nt * 16;
                v16bf Bv;
#pragma unroll
                for (int j = 0; j < 8; ++j) {
                    Bv[j]     = (bf16_t)bp[(size_t)(8 * g + j) * D_DIM];
                    Bv[j + 8] = (bf16_t)bp[(size_t)(16 + 8 * g + j) * D_DIM];
                }
                o[nt] = __builtin_amdgcn_wmma_f32_16x16x32_bf16(false, A, false, Bv,
                                                                (short)0, o[nt],
                                                                false, false);
            }
        }
    }

    int tk[8]; float tw[8];
#pragma unroll
    for (int v = 0; v < 8; ++v) { tk[v] = sTok[v + 8 * g]; tw[v] = sWt[v + 8 * g]; }
#pragma unroll
    for (int nt = 0; nt < 8; ++nt)
#pragma unroll
        for (int v = 0; v < 8; ++v)
            atomicAdd(out + (size_t)tk[v] * D_DIM + oc0 + nt * 16 + lo,
                      o[nt][v] * tw[v]);
}

// ===========================================================================
// Host launcher
// ===========================================================================
extern "C" void kernel_launch(void* const* d_in, const int* in_sizes, int n_in,
                              void* d_out, int out_size, void* d_ws, size_t ws_size,
                              hipStream_t stream)
{
    const float* x  = (const float*)d_in[0];   // [T, D]
    const float* gw = (const float*)d_in[1];   // [D, E]
    const float* w1 = (const float*)d_in[2];   // [E, D, H]
    const float* w2 = (const float*)d_in[3];   // [E, H, D]

    float* out    = (float*)d_out;                   // [T, D]
    float* logits = out + (size_t)T_TOK * D_DIM;     // [T, E]

    char* ws = (char*)d_ws;
    int*   sel     = (int*)  (ws);                    //  64 KB
    float* wts     = (float*)(ws + (64u  << 10));     //  64 KB
    int*   listTok = (int*)  (ws + (128u << 10));     // 256 KB
    float* listWt  = (float*)(ws + (384u << 10));     // 256 KB
    int*   ntile   = (int*)  (ws + (640u << 10));     // 16 ints
    bf16_t* w1s    = (bf16_t*)(ws + (1u << 20));                       // 64 MB
    bf16_t* w2s    = (bf16_t*)(ws + (1u << 20) + ((size_t)64 << 20));  // 64 MB
    const size_t ws_need = (1u << 20) + ((size_t)128 << 20);

    router_kernel<<<dim3(T_TOK / 8), dim3(256), 0, stream>>>(x, gw, logits, sel, wts);
    build_lists_kernel<<<dim3(1), dim3(256), 0, stream>>>(sel, wts, listTok, listWt, ntile);
    zero_kernel<<<dim3((T_TOK * D_DIM / 4) / 256), dim3(256), 0, stream>>>(
        (float4*)out, T_TOK * D_DIM / 4);

    if (ws_size >= ws_need) {
        // one-time fp32 -> bf16 WMMA-operand-order swizzle (coalesced reads)
        swz_w1_kernel<<<dim3(65536 / 8), dim3(256), 0, stream>>>(w1, w1s);
        swz_w2_kernel<<<dim3(65536 / 8), dim3(256), 0, stream>>>(w2, w2s);
        // expert-major grid: per-expert 16 MB bf16 weights stay L2-resident
        moe_ffn32_kernel<<<dim3(T_TOK / 32, E_NUM), dim3(512), 0, stream>>>(
            x, w1s, w2s, listTok, listWt, ntile, out);
    } else {
        moe_ffn_kernel<<<dim3(T_TOK / 16, E_NUM), dim3(256), 0, stream>>>(
            x, w1, w2, listTok, listWt, ntile, out);
    }
}
